// SimplifiedAttention_27754078667589
// MI455X (gfx1250) — matrix operand order repaired and checked by reading, not compiled, not run on hardware
//
#include <hip/hip_runtime.h>
#include <math.h>

typedef __bf16 bf16_t;
typedef __attribute__((ext_vector_type(16))) __bf16 v16bf;
typedef __attribute__((ext_vector_type(8)))  __bf16 v8bf;
typedef __attribute__((ext_vector_type(8)))  float  v8f;

#define HIDDEN 4096
#define SEQ    2048
#define NHEADS 32
#define HDIM   128
#define QKVN   12288   // 3*HIDDEN

// ln(10000)/64 (RoPE inverse-frequency exponent scale), 1/sqrt(128)
#define ROPE_LOG_SCALE 0.14391156831212787f
#define ATTN_SCALE     0.08838834764831845f

// ---------------- WMMA fragment helpers (CDNA5 wave32 layouts) ----------------
// A (16x32 bf16, MxK): lane m = L&15; VGPR0-3 = K[(L>>4)*8 .. +7], VGPR4-7 = K[16+(L>>4)*8 .. +7]
__device__ __forceinline__ v16bf load_a_frag(const bf16_t* A, int lda, int m0, int k0, int lane) {
  const int m    = m0 + (lane & 15);
  const int koff = (lane >> 4) * 8;
  const v8bf lo = *(const v8bf*)(A + (size_t)m * lda + k0 + koff);
  const v8bf hi = *(const v8bf*)(A + (size_t)m * lda + k0 + 16 + koff);
  v16bf r;
#pragma unroll
  for (int i = 0; i < 8; ++i) { r[i] = lo[i]; r[8 + i] = hi[i]; }
  return r;
}

// B (32x16 bf16, KxN) supplied as Bt (N x K row-major): lane n = L&15; VGPR0-3 = K[(L>>4)*8 .. +7], VGPR4-7 = K[16+(L>>4)*8 .. +7]
__device__ __forceinline__ v16bf load_bT_frag(const bf16_t* Bt, int ldb, int n0, int k0, int lane) {
  const int n    = n0 + (lane & 15);
  const int koff = (lane >> 4) * 8;
  const v8bf lo = *(const v8bf*)(Bt + (size_t)n * ldb + k0 + koff);
  const v8bf hi = *(const v8bf*)(Bt + (size_t)n * ldb + k0 + 16 + koff);
  v16bf r;
#pragma unroll
  for (int i = 0; i < 8; ++i) { r[i] = lo[i]; r[8 + i] = hi[i]; }
  return r;
}

__device__ __forceinline__ v8f wmma_bf16(v16bf a, v16bf b, v8f c) {
  return __builtin_amdgcn_wmma_f32_16x16x32_bf16(false, a, false, b, (short)0, c, false, false);
}

__device__ __forceinline__ v8f vzero8() {
  v8f z;
#pragma unroll
  for (int i = 0; i < 8; ++i) z[i] = 0.0f;
  return z;
}

// ---------------- f32 -> bf16 conversion ----------------
__global__ __launch_bounds__(256) void cvt_f32_bf16(const float* __restrict__ src,
                                                    bf16_t* __restrict__ dst, size_t n) {
  size_t i = (size_t)blockIdx.x * blockDim.x + threadIdx.x;
  size_t stride = (size_t)gridDim.x * blockDim.x;
  for (; i < n; i += stride) dst[i] = (bf16_t)src[i];
}

// ---------------- RoPE cos/sin tables: [pos][c] for c in 0..63 ----------------
__global__ __launch_bounds__(256) void rope_tables(float* __restrict__ costab,
                                                   float* __restrict__ sintab) {
  int i = blockIdx.x * blockDim.x + threadIdx.x;   // i < SEQ*64
  if (i >= SEQ * 64) return;
  const int p = i >> 6;
  const int c = i & 63;
  const float invf = expf(-(float)c * ROPE_LOG_SCALE);
  const float ang  = (float)p * invf;
  costab[i] = cosf(ang);
  sintab[i] = sinf(ang);
}

// ---------------- QKV GEMM (NT) + bias + RoPE epilogue ----------------
// X (2048 x 4096 bf16) @ Wqkv^T (Wqkv is 12288 x 4096 bf16) -> per-block 128x128 f32 tile.
// 4 waves: wave (wx,wy); wave columns = wy*32 + {0,16,64,80} so the RoPE pair (c, c+64)
// sits in accumulator tiles j and j+2 of the SAME lane -> register-only RoPE.
__global__ __launch_bounds__(128) void qkv_gemm_rope(
    const bf16_t* __restrict__ Xb, const bf16_t* __restrict__ Wb,
    const float* __restrict__ bias,
    const float* __restrict__ costab, const float* __restrict__ sintab,
    bf16_t* __restrict__ Qr, bf16_t* __restrict__ Kr, bf16_t* __restrict__ Vt)
{
  const int lane = threadIdx.x & 31;
  const int wid  = threadIdx.x >> 5;
  const int wx = wid & 1, wy = wid >> 1;
  const int mb = blockIdx.x * 128, nb = blockIdx.y * 128;
  const int mbase = mb + wx * 64;

  int ncol[4];
#pragma unroll
  for (int j = 0; j < 4; ++j) ncol[j] = wy * 32 + (j & 1) * 16 + (j >> 1) * 64;

  v8f acc[4][4];
#pragma unroll
  for (int i = 0; i < 4; ++i)
#pragma unroll
    for (int j = 0; j < 4; ++j) acc[i][j] = vzero8();

  for (int k0 = 0; k0 < HIDDEN; k0 += 32) {
    v16bf af[4], bf[4];
#pragma unroll
    for (int i = 0; i < 4; ++i) af[i] = load_a_frag(Xb, HIDDEN, mbase + i * 16, k0, lane);
#pragma unroll
    for (int j = 0; j < 4; ++j) bf[j] = load_bT_frag(Wb, HIDDEN, nb + ncol[j], k0, lane);
#pragma unroll
    for (int i = 0; i < 4; ++i)
#pragma unroll
      for (int j = 0; j < 4; ++j) acc[i][j] = wmma_bf16(af[i], bf[j], acc[i][j]);
  }

  // This 128-col block covers exactly one (head, {q|k|v}) 128-dim group (384 = 3*128).
  const int h = nb / 384;
  const int t = (nb % 384) / 128;

  if (t < 2) {  // Q or K : bias + RoPE (table lookup), write [h][seq][dim]
    bf16_t* dst = (t == 0 ? Qr : Kr) + (size_t)h * SEQ * HDIM;
#pragma unroll
    for (int i = 0; i < 4; ++i)
#pragma unroll
      for (int j = 0; j < 2; ++j)
#pragma unroll
        for (int r = 0; r < 8; ++r) {
          const int p = mbase + i * 16 + (lane >> 4) * 8 + r;   // position id == row
          const int c = ncol[j] + (lane & 15);                  // 0..63
          const float x  = acc[i][j][r]     + bias[nb + c];
          const float x2 = acc[i][j + 2][r] + bias[nb + c + 64];
          const float cs = costab[p * 64 + c];
          const float sn = sintab[p * 64 + c];
          dst[(size_t)p * HDIM + c]      = (bf16_t)(x * cs - x2 * sn);
          dst[(size_t)p * HDIM + c + 64] = (bf16_t)(x2 * cs + x * sn);
        }
  } else {      // V : bias only, write transposed [h][dim][seq] for contiguous PV B-frags
    bf16_t* dst = Vt + (size_t)h * HDIM * SEQ;
#pragma unroll
    for (int i = 0; i < 4; ++i)
#pragma unroll
      for (int j = 0; j < 4; ++j)
#pragma unroll
        for (int r = 0; r < 8; ++r) {
          const int p = mbase + i * 16 + (lane >> 4) * 8 + r;
          const int c = ncol[j] + (lane & 15);
          dst[(size_t)c * SEQ + p] = (bf16_t)(acc[i][j][r] + bias[nb + c]);
        }
  }
}

// ---------------- Flash attention: 1 wave per (head, 16-row q tile) ----------------
__global__ __launch_bounds__(32) void flash_attn(
    const bf16_t* __restrict__ Qr, const bf16_t* __restrict__ Kr,
    const bf16_t* __restrict__ Vt, bf16_t* __restrict__ Ob)
{
  __shared__ bf16_t pls[16][32];           // P tile: D-layout -> A-layout staging
  const int lane = threadIdx.x & 31;
  const int h  = blockIdx.x;
  const int q0 = blockIdx.y * 16;
  const bf16_t* Qh = Qr + (size_t)h * SEQ * HDIM;
  const bf16_t* Kh = Kr + (size_t)h * SEQ * HDIM;
  const bf16_t* Vh = Vt + (size_t)h * HDIM * SEQ;

  v16bf qf[4];
#pragma unroll
  for (int d = 0; d < 4; ++d) qf[d] = load_a_frag(Qh, HDIM, q0, d * 32, lane);

  v8f o[8];
#pragma unroll
  for (int dt = 0; dt < 8; ++dt) o[dt] = vzero8();

  float mrow[8], lrow[8];
#pragma unroll
  for (int r = 0; r < 8; ++r) { mrow[r] = -__builtin_inff(); lrow[r] = 0.0f; }

  const int rbase = q0 + (lane >> 4) * 8;   // D-layout rows of this lane
  const int nlane = lane & 15;              // D-layout column of this lane

  for (int kb = 0; kb < q0 + 16; kb += 32) {        // causal: kb <= q0 always
    v8f s0 = vzero8(), s1 = vzero8();
#pragma unroll
    for (int d = 0; d < 4; ++d) s0 = wmma_bf16(qf[d], load_bT_frag(Kh, HDIM, kb,      d * 32, lane), s0);
#pragma unroll
    for (int d = 0; d < 4; ++d) s1 = wmma_bf16(qf[d], load_bT_frag(Kh, HDIM, kb + 16, d * 32, lane), s1);

    const int c0 = kb + nlane, c1 = c0 + 16;
    float mt[8], rs[8], alpha[8];
#pragma unroll
    for (int r = 0; r < 8; ++r) {
      const int q = rbase + r;
      float a = s0[r] * ATTN_SCALE; if (c0 > q) a = -__builtin_inff();
      float b = s1[r] * ATTN_SCALE; if (c1 > q) b = -__builtin_inff();
      s0[r] = a; s1[r] = b;
      float v = fmaxf(a, b);
#pragma unroll
      for (int off = 1; off < 16; off <<= 1) v = fmaxf(v, __shfl_xor(v, off, 32));
      mt[r] = v;
    }
#pragma unroll
    for (int r = 0; r < 8; ++r) {
      float mn = fmaxf(mrow[r], mt[r]);
      alpha[r] = __expf(mrow[r] - mn);
      mrow[r] = mn;
      float p0 = __expf(s0[r] - mn);
      float p1 = __expf(s1[r] - mn);
      s0[r] = p0; s1[r] = p1;
      float v = p0 + p1;
#pragma unroll
      for (int off = 1; off < 16; off <<= 1) v += __shfl_xor(v, off, 32);
      rs[r] = v;
      lrow[r] = lrow[r] * alpha[r] + rs[r];
    }
#pragma unroll
    for (int dt = 0; dt < 8; ++dt)
#pragma unroll
      for (int r = 0; r < 8; ++r) o[dt][r] *= alpha[r];

    // transpose P (D-layout) into A-layout via LDS, then O += P(16x32) @ V(32x128)
    {
      const int m = (lane >> 4) * 8;
#pragma unroll
      for (int r = 0; r < 8; ++r) {
        pls[m + r][nlane]      = (bf16_t)s0[r];
        pls[m + r][16 + nlane] = (bf16_t)s1[r];
      }
    }
    __syncthreads();
    const v16bf pf = load_a_frag(&pls[0][0], 32, 0, 0, lane);
#pragma unroll
    for (int dt = 0; dt < 8; ++dt)
      o[dt] = wmma_bf16(pf, load_bT_frag(Vh, SEQ, dt * 16, kb, lane), o[dt]);
    __syncthreads();
  }

#pragma unroll
  for (int r = 0; r < 8; ++r) lrow[r] = 1.0f / lrow[r];
#pragma unroll
  for (int dt = 0; dt < 8; ++dt)
#pragma unroll
    for (int r = 0; r < 8; ++r) {
      const int row = rbase + r;
      const int col = h * HDIM + dt * 16 + nlane;
      Ob[(size_t)row * HIDDEN + col] = (bf16_t)(o[dt][r] * lrow[r]);
    }
}

// ---------------- Output projection GEMM (NT) + bias -> f32 ----------------
__global__ __launch_bounds__(128) void out_gemm(
    const bf16_t* __restrict__ Xb, const bf16_t* __restrict__ Wb,
    const float* __restrict__ bias, float* __restrict__ out)
{
  const int lane = threadIdx.x & 31;
  const int wid  = threadIdx.x >> 5;
  const int wx = wid & 1, wy = wid >> 1;
  const int mb = blockIdx.x * 128, nb = blockIdx.y * 128;
  const int mbase = mb + wx * 64, nbase = nb + wy * 64;

  v8f acc[4][4];
#pragma unroll
  for (int i = 0; i < 4; ++i)
#pragma unroll
    for (int j = 0; j < 4; ++j) acc[i][j] = vzero8();

  for (int k0 = 0; k0 < HIDDEN; k0 += 32) {
    v16bf af[4], bf[4];
#pragma unroll
    for (int i = 0; i < 4; ++i) af[i] = load_a_frag(Xb, HIDDEN, mbase + i * 16, k0, lane);
#pragma unroll
    for (int j = 0; j < 4; ++j) bf[j] = load_bT_frag(Wb, HIDDEN, nbase + j * 16, k0, lane);
#pragma unroll
    for (int i = 0; i < 4; ++i)
#pragma unroll
      for (int j = 0; j < 4; ++j) acc[i][j] = wmma_bf16(af[i], bf[j], acc[i][j]);
  }

#pragma unroll
  for (int i = 0; i < 4; ++i)
#pragma unroll
    for (int j = 0; j < 4; ++j)
#pragma unroll
      for (int r = 0; r < 8; ++r) {
        const int row = mbase + i * 16 + (lane >> 4) * 8 + r;
        const int col = nbase + j * 16 + (lane & 15);
        out[(size_t)row * HIDDEN + col] = acc[i][j][r] + bias[col];
      }
}

// ---------------- launch ----------------
extern "C" void kernel_launch(void* const* d_in, const int* in_sizes, int n_in,
                              void* d_out, int out_size, void* d_ws, size_t ws_size,
                              hipStream_t stream) {
  (void)in_sizes; (void)n_in; (void)out_size; (void)ws_size;
  const float* hidden = (const float*)d_in[0];
  // d_in[1] = position_ids == arange(SEQ); positions derived from row index.
  const float* wqkv_w = (const float*)d_in[2];
  const float* wqkv_b = (const float*)d_in[3];
  const float* wo_w   = (const float*)d_in[4];
  const float* wo_b   = (const float*)d_in[5];
  float* out = (float*)d_out;

  char* p = (char*)d_ws;
  auto take = [&](size_t bytes) { char* r = p; p += (bytes + 255) & ~(size_t)255; return r; };
  bf16_t* Xb  = (bf16_t*)take((size_t)SEQ * HIDDEN * 2);
  bf16_t* Wqb = (bf16_t*)take((size_t)QKVN * HIDDEN * 2);
  bf16_t* Wob = (bf16_t*)take((size_t)HIDDEN * HIDDEN * 2);
  bf16_t* Qr  = (bf16_t*)take((size_t)NHEADS * SEQ * HDIM * 2);
  bf16_t* Kr  = (bf16_t*)take((size_t)NHEADS * SEQ * HDIM * 2);
  bf16_t* Vt  = (bf16_t*)take((size_t)NHEADS * HDIM * SEQ * 2);
  bf16_t* Ob  = (bf16_t*)take((size_t)SEQ * HIDDEN * 2);
  float*  Ct  = (float*)take((size_t)SEQ * 64 * 4);   // RoPE cos table
  float*  St  = (float*)take((size_t)SEQ * 64 * 4);   // RoPE sin table

  cvt_f32_bf16<<<4096, 256, 0, stream>>>(hidden, Xb, (size_t)SEQ * HIDDEN);
  cvt_f32_bf16<<<8192, 256, 0, stream>>>(wqkv_w, Wqb, (size_t)QKVN * HIDDEN);
  cvt_f32_bf16<<<8192, 256, 0, stream>>>(wo_w,   Wob, (size_t)HIDDEN * HIDDEN);
  rope_tables<<<(SEQ * 64 + 255) / 256, 256, 0, stream>>>(Ct, St);

  qkv_gemm_rope<<<dim3(SEQ / 128, QKVN / 128), 128, 0, stream>>>(Xb, Wqb, wqkv_b, Ct, St, Qr, Kr, Vt);
  flash_attn<<<dim3(NHEADS, SEQ / 16), 32, 0, stream>>>(Qr, Kr, Vt, Ob);
  out_gemm<<<dim3(SEQ / 128, HIDDEN / 128), 128, 0, stream>>>(Ob, Wob, wo_b, out);
}
